// GAT_4776003633538
// MI455X (gfx1250) — matrix operand order repaired and checked by reading, not compiled, not run on hardware
//
#include <hip/hip_runtime.h>
#include <hip/hip_bf16.h>

// ---------------------------------------------------------------------------
// Attention block: out = gelu_exact(softmax(Q K^T) V) + x
//   x  : [4, 4096, 1024] f32,  Wk/Wq/Wv : [1024, 1024] f32
// MI455X / gfx1250 (wave32): all matmuls on v_wmma_f32_16x16x32_bf16 with
// f32 accumulate. Shared B tiles are DOUBLE-BUFFERED in LDS and filled with
// global_load_async_to_lds_b128 (ASYNCcnt), overlapping DMA with WMMA.
// Each wave owns a 32-row M tile (two A fragments), so every B fragment
// staged through LDS feeds TWO WMMAs -> half the ds/barrier cost per FLOP.
//
// Pipeline:
//   K1: convert+transpose weights -> W^T bf16 (LDS tile transpose, z=3 fused)
//   K2: projection GEMM (WMMA): Q,K row-major bf16 ; V stored transposed [D][S]
//   per batch:
//   K3: S = Q K^T (WMMA) -> f32 [4096,4096]
//   K4: row softmax (wave32 shuffle reduce) -> P bf16
//   K5: O = P V (WMMA) + exact GELU + residual -> f32 out (non-temporal)
//
// Workspace layout (bytes): needs ~198 MB
//   [0,2M)  WqT   [2M,4M) WkT   [4M,6M) WvT
//   [6M,38M)   Q bf16 [4][4096][1024]
//   [38M,70M)  K bf16 [4][4096][1024]
//   [70M,102M) Vt bf16 [4][1024][4096]
//   [102M,166M) S f32 [4096][4096]   (reused per batch)
//   [166M,198M) P bf16 [4096][4096]  (reused per batch)
// ---------------------------------------------------------------------------

typedef __attribute__((ext_vector_type(16))) __bf16 v16bf;
typedef __attribute__((ext_vector_type(8)))  float  v8f;
typedef int vint4 __attribute__((vector_size(16)));   // b128 payload type

#define SEQ    4096
#define DIM    1024
#define NBATCH 4
#define KP     40     // padded LDS row stride (elements): 80B -> conflict-free b128

#if __has_builtin(__builtin_amdgcn_global_load_async_to_lds_b128) && \
    __has_builtin(__builtin_amdgcn_s_wait_asynccnt)
#define USE_ASYNC_LDS 1
typedef __attribute__((address_space(1))) vint4* gb128_t;
typedef __attribute__((address_space(3))) vint4* lb128_t;
#endif

// A-fragment (16x32 bf16): lane needs two 16B chunks: [k0..k0+7], [k0+16..k0+23]
__device__ __forceinline__ v16bf load_a_bf16(const __bf16* p0, const __bf16* p1) {
    union { uint4 u[2]; v16bf v; } r;
    r.u[0] = *reinterpret_cast<const uint4*>(p0);
    r.u[1] = *reinterpret_cast<const uint4*>(p1);
    return r.v;
}

// B-fragment (32x16 bf16): lane needs two 16B chunks: [k0..k0+7], [k0+16..k0+23] -> 2x ds b128
__device__ __forceinline__ v16bf load_b_frag(const __bf16* p) {
    union { uint4 u[2]; v16bf v; } r;
    r.u[0] = *reinterpret_cast<const uint4*>(p);
    r.u[1] = *reinterpret_cast<const uint4*>(p + 16);
    return r.v;
}

// A-fragment sourced from f32 (convert on the fly)
__device__ __forceinline__ v16bf cvt_a_f32(const float* p0, const float* p1) {
    v16bf a;
#pragma unroll
    for (int e = 0; e < 8; ++e) a[e] = (__bf16)p0[e];
#pragma unroll
    for (int e = 0; e < 8; ++e) a[8 + e] = (__bf16)p1[e];
    return a;
}

__device__ __forceinline__ v8f wmma_bf16(v16bf a, v16bf b, v8f c) {
    return __builtin_amdgcn_wmma_f32_16x16x32_bf16(
        /*neg_a=*/false, a, /*neg_b=*/false, b,
        /*c_mod=*/(short)0, c, /*reuse_a=*/false, /*reuse_b=*/false);
}

// Stage an NT x 32 bf16 B-tile (rows of `src`, columns [kk,kk+32)) into LDS
// with padded row stride KP. 256 threads; each moves (NT*4/256) 16B chunks.
template <int NT>
__device__ __forceinline__ void stage_b_tile(__bf16* lds, const __bf16* __restrict__ src,
                                             int row_stride, int kk) {
    constexpr int PT = (NT * 4) / 256;          // 16B chunks per thread
#pragma unroll
    for (int i = 0; i < PT; ++i) {
        int c = threadIdx.x + i * 256;
        int n = c >> 2;
        int part = c & 3;
        const __bf16* g = src + (size_t)n * row_stride + kk + part * 8;
        __bf16* l = lds + n * KP + part * 8;
#if defined(USE_ASYNC_LDS)
        __builtin_amdgcn_global_load_async_to_lds_b128((gb128_t)g, (lb128_t)l, 0, 0);
#else
        *reinterpret_cast<uint4*>(l) = *reinterpret_cast<const uint4*>(g);
#endif
    }
}

// Publish the most recently staged tile to all waves of the block.
__device__ __forceinline__ void stage_publish() {
#if defined(USE_ASYNC_LDS)
    __builtin_amdgcn_s_wait_asynccnt(0);
#endif
    __syncthreads();
}

// ---------------- K1: W [d][n] f32 -> Wt [n][d] bf16 (LDS transpose) --------
__global__ __launch_bounds__(256) void wcvt3_kernel(const float* __restrict__ wq,
                                                    const float* __restrict__ wk,
                                                    const float* __restrict__ wv,
                                                    __bf16* __restrict__ wqt,
                                                    __bf16* __restrict__ wkt,
                                                    __bf16* __restrict__ wvt) {
    __shared__ float tile[32][33];
    const float* w = (blockIdx.z == 0) ? wq : (blockIdx.z == 1) ? wk : wv;
    __bf16*     wt = (blockIdx.z == 0) ? wqt : (blockIdx.z == 1) ? wkt : wvt;
    const int bn = blockIdx.x * 32;   // output row block (n)
    const int bd = blockIdx.y * 32;   // output col block (d)
    const int tx = threadIdx.x & 31;
    const int ty = threadIdx.x >> 5;  // 0..7
#pragma unroll
    for (int i = 0; i < 32; i += 8)
        tile[ty + i][tx] = w[(size_t)(bd + ty + i) * DIM + bn + tx];
    __syncthreads();
#pragma unroll
    for (int i = 0; i < 32; i += 8)
        wt[(size_t)(bn + ty + i) * DIM + bd + tx] = (__bf16)tile[tx][ty + i];
}

// ---------------- K2: projections ------------------------------------------
// grid (64, 16); block 256 (8 waves). Wave tile: 32 rows x 64 cols.
// out_rm != null -> row-major bf16 [16384][1024] (Q or K)
// else           -> transposed bf16 per batch [4][1024][4096] (Vt)
__global__ __launch_bounds__(256) void proj_kernel(const float*  __restrict__ x,
                                                   const __bf16* __restrict__ wt,
                                                   __bf16* __restrict__ out_rm,
                                                   __bf16* __restrict__ out_tr) {
    __shared__ __bf16 bt[2][64 * KP];
    const int lane = threadIdx.x & 31;
    const int wave = threadIdx.x >> 5;
    const int m0   = blockIdx.x * 256 + wave * 32;   // global row (b*4096+s)
    const int n0   = blockIdx.y * 64;

    const int mA = lane & 15;
    const int kA = (lane < 16) ? 0 : 8;
    const int nB = lane & 15;
    const int kB = (lane < 16) ? 0 : 8;

    v8f acc0[4], acc1[4];
#pragma unroll
    for (int f = 0; f < 4; ++f) { acc0[f] = (v8f){}; acc1[f] = (v8f){}; }

    const float*  xrow0 = x + (size_t)(m0 + mA) * DIM;
    const float*  xrow1 = x + (size_t)(m0 + 16 + mA) * DIM;
    const __bf16* wbase = wt + (size_t)n0 * DIM;

    constexpr int NIT = DIM / 32;
    stage_b_tile<64>(bt[0], wbase, DIM, 0);
    for (int it = 0; it < NIT; ++it) {
        const int kk = it * 32;
        stage_publish();                              // tile[it] ready everywhere
        const __bf16* cur = bt[it & 1];
        if (it + 1 < NIT) {                           // overlap DMA with WMMA
            stage_b_tile<64>(bt[(it + 1) & 1], wbase, DIM, kk + 32);
            __builtin_prefetch(xrow0 + kk + 32);
            __builtin_prefetch(xrow1 + kk + 32);
        }
        v16bf a0 = cvt_a_f32(xrow0 + kk + kA, xrow0 + kk + kA + 16);
        v16bf a1 = cvt_a_f32(xrow1 + kk + kA, xrow1 + kk + kA + 16);
#pragma unroll
        for (int f = 0; f < 4; ++f) {
            v16bf b = load_b_frag(cur + (f * 16 + nB) * KP + kB);
            acc0[f] = wmma_bf16(a0, b, acc0[f]);
            acc1[f] = wmma_bf16(a1, b, acc1[f]);
        }
    }

    const int mC = (lane < 16) ? 0 : 8;
    if (out_rm) {
#pragma unroll
        for (int f = 0; f < 4; ++f) {
            int n = n0 + f * 16 + (lane & 15);
#pragma unroll
            for (int r = 0; r < 8; ++r) {
                out_rm[(size_t)(m0 + mC + r) * DIM + n]      = (__bf16)acc0[f][r];
                out_rm[(size_t)(m0 + 16 + mC + r) * DIM + n] = (__bf16)acc1[f][r];
            }
        }
    } else {
        int b  = m0 >> 12;
        int s0 = (m0 & 4095) + mC;             // contiguous in M -> packed store
#pragma unroll
        for (int f = 0; f < 4; ++f) {
            int n = n0 + f * 16 + (lane & 15);
            union { __bf16 h[8]; uint4 u; } pk0, pk1;
#pragma unroll
            for (int r = 0; r < 8; ++r) { pk0.h[r] = (__bf16)acc0[f][r];
                                          pk1.h[r] = (__bf16)acc1[f][r]; }
            __bf16* base = out_tr + (size_t)b * DIM * SEQ + (size_t)n * SEQ + s0;
            *reinterpret_cast<uint4*>(base)      = pk0.u;
            *reinterpret_cast<uint4*>(base + 16) = pk1.u;
        }
    }
}

// ---------------- K3: S = Q K^T (one batch) --------------------------------
// grid (16, 32); block 256. Wave tile: 32 q x 128 keys (16 C fragments).
__global__ __launch_bounds__(256) void qk_kernel(const __bf16* __restrict__ q,
                                                 const __bf16* __restrict__ k,
                                                 float* __restrict__ s) {
    __shared__ __bf16 bt[2][128 * KP];
    const int lane = threadIdx.x & 31;
    const int wave = threadIdx.x >> 5;
    const int q0 = blockIdx.x * 256 + wave * 32;
    const int n0 = blockIdx.y * 128;

    const int mA = lane & 15;
    const int kA = (lane < 16) ? 0 : 8;
    const int nB = lane & 15;
    const int kB = (lane < 16) ? 0 : 8;

    v8f acc0[8], acc1[8];
#pragma unroll
    for (int f = 0; f < 8; ++f) { acc0[f] = (v8f){}; acc1[f] = (v8f){}; }

    const __bf16* qrow0 = q + (size_t)(q0 + mA) * DIM;
    const __bf16* qrow1 = q + (size_t)(q0 + 16 + mA) * DIM;
    const __bf16* kbase = k + (size_t)n0 * DIM;

    constexpr int NIT = DIM / 32;
    stage_b_tile<128>(bt[0], kbase, DIM, 0);
    for (int it = 0; it < NIT; ++it) {
        const int kk = it * 32;
        stage_publish();
        const __bf16* cur = bt[it & 1];
        if (it + 1 < NIT) {
            stage_b_tile<128>(bt[(it + 1) & 1], kbase, DIM, kk + 32);
            __builtin_prefetch(qrow0 + kk + 32);
            __builtin_prefetch(qrow1 + kk + 32);
        }
        v16bf a0 = load_a_bf16(qrow0 + kk + kA, qrow0 + kk + kA + 16);
        v16bf a1 = load_a_bf16(qrow1 + kk + kA, qrow1 + kk + kA + 16);
#pragma unroll
        for (int f = 0; f < 8; ++f) {
            v16bf b = load_b_frag(cur + (f * 16 + nB) * KP + kB);
            acc0[f] = wmma_bf16(a0, b, acc0[f]);
            acc1[f] = wmma_bf16(a1, b, acc1[f]);
        }
    }

    const int mC = (lane < 16) ? 0 : 8;
#pragma unroll
    for (int f = 0; f < 8; ++f) {
        int n = n0 + f * 16 + (lane & 15);
#pragma unroll
        for (int r = 0; r < 8; ++r) {
            s[(size_t)(q0 + mC + r) * SEQ + n]      = acc0[f][r];
            s[(size_t)(q0 + 16 + mC + r) * SEQ + n] = acc1[f][r];
        }
    }
}

// ---------------- K4: row softmax, f32 S -> bf16 P -------------------------
// wave32 shuffle reduction + one 8-partial LDS pass (2 barriers total).
__global__ __launch_bounds__(256) void softmax_kernel(const float* __restrict__ s,
                                                      __bf16* __restrict__ p) {
    __shared__ float red[16];                  // [0..7] max, [8..15] sum
    const int lane = threadIdx.x & 31;
    const int wave = threadIdx.x >> 5;
    const int row = blockIdx.x;
    const float* sr = s + (size_t)row * SEQ;

    float vals[16];
    float mx = -3.402823466e38f;
#pragma unroll
    for (int i = 0; i < 16; ++i) {
        vals[i] = sr[threadIdx.x + i * 256];
        mx = fmaxf(mx, vals[i]);
    }
#pragma unroll
    for (int off = 16; off > 0; off >>= 1)
        mx = fmaxf(mx, __shfl_xor(mx, off, 32));
    if (lane == 0) red[wave] = mx;
    __syncthreads();
    float m = red[0];
#pragma unroll
    for (int w = 1; w < 8; ++w) m = fmaxf(m, red[w]);

    float sum = 0.0f;
#pragma unroll
    for (int i = 0; i < 16; ++i) {
        vals[i] = expf(vals[i] - m);
        sum += vals[i];
    }
#pragma unroll
    for (int off = 16; off > 0; off >>= 1)
        sum += __shfl_xor(sum, off, 32);
    if (lane == 0) red[8 + wave] = sum;
    __syncthreads();
    float t = 0.0f;
#pragma unroll
    for (int w = 0; w < 8; ++w) t += red[8 + w];
    float inv = 1.0f / t;

    __bf16* pr = p + (size_t)row * SEQ;
#pragma unroll
    for (int i = 0; i < 16; ++i)
        pr[threadIdx.x + i * 256] = (__bf16)(vals[i] * inv);
}

// ---------------- K5: O = P V, + exact GELU + residual ---------------------
// grid (16, 16); block 256. Wave tile: 32 q x 64 dims.
__global__ __launch_bounds__(256) void pv_kernel(const __bf16* __restrict__ p,
                                                 const __bf16* __restrict__ vt,
                                                 const float*  __restrict__ xb,
                                                 float* __restrict__ out) {
    __shared__ __bf16 bt[2][64 * KP];
    const int lane = threadIdx.x & 31;
    const int wave = threadIdx.x >> 5;
    const int q0 = blockIdx.x * 256 + wave * 32;
    const int n0 = blockIdx.y * 64;

    const int mA = lane & 15;
    const int kA = (lane < 16) ? 0 : 8;
    const int nB = lane & 15;
    const int kB = (lane < 16) ? 0 : 8;

    v8f acc0[4], acc1[4];
#pragma unroll
    for (int f = 0; f < 4; ++f) { acc0[f] = (v8f){}; acc1[f] = (v8f){}; }

    const __bf16* prow0 = p + (size_t)(q0 + mA) * SEQ;
    const __bf16* prow1 = p + (size_t)(q0 + 16 + mA) * SEQ;
    const __bf16* vbase = vt + (size_t)n0 * SEQ;

    constexpr int NIT = SEQ / 32;
    stage_b_tile<64>(bt[0], vbase, SEQ, 0);
    for (int it = 0; it < NIT; ++it) {
        const int kk = it * 32;
        stage_publish();
        const __bf16* cur = bt[it & 1];
        if (it + 1 < NIT) {
            stage_b_tile<64>(bt[(it + 1) & 1], vbase, SEQ, kk + 32);
            __builtin_prefetch(prow0 + kk + 32);
            __builtin_prefetch(prow1 + kk + 32);
        }
        v16bf a0 = load_a_bf16(prow0 + kk + kA, prow0 + kk + kA + 16);
        v16bf a1 = load_a_bf16(prow1 + kk + kA, prow1 + kk + kA + 16);
#pragma unroll
        for (int f = 0; f < 4; ++f) {
            v16bf b = load_b_frag(cur + (f * 16 + nB) * KP + kB);
            acc0[f] = wmma_bf16(a0, b, acc0[f]);
            acc1[f] = wmma_bf16(a1, b, acc1[f]);
        }
    }

    const int mC = (lane < 16) ? 0 : 8;
#pragma unroll
    for (int f = 0; f < 4; ++f) {
        int n = n0 + f * 16 + (lane & 15);
#pragma unroll
        for (int r = 0; r < 8; ++r) {
            size_t i0 = (size_t)(q0 + mC + r) * DIM + n;
            size_t i1 = (size_t)(q0 + 16 + mC + r) * DIM + n;
            float v0 = acc0[f][r], v1 = acc1[f][r];
            float g0 = 0.5f * v0 * (1.0f + erff(v0 * 0.70710678118654752f));
            float g1 = 0.5f * v1 * (1.0f + erff(v1 * 0.70710678118654752f));
            __builtin_nontemporal_store(g0 + xb[i0], &out[i0]);
            __builtin_nontemporal_store(g1 + xb[i1], &out[i1]);
        }
    }
}

// ---------------------------------------------------------------------------
extern "C" void kernel_launch(void* const* d_in, const int* in_sizes, int n_in,
                              void* d_out, int out_size, void* d_ws, size_t ws_size,
                              hipStream_t stream) {
    const float* x  = (const float*)d_in[0];
    const float* Wk = (const float*)d_in[1];
    const float* Wq = (const float*)d_in[2];
    const float* Wv = (const float*)d_in[3];
    float* out = (float*)d_out;

    char* ws = (char*)d_ws;
    const size_t MB = 1024ull * 1024ull;
    __bf16* WqT = (__bf16*)(ws + 0 * MB);
    __bf16* WkT = (__bf16*)(ws + 2 * MB);
    __bf16* WvT = (__bf16*)(ws + 4 * MB);
    __bf16* Qb  = (__bf16*)(ws + 6 * MB);
    __bf16* Kb  = (__bf16*)(ws + 38 * MB);
    __bf16* Vt  = (__bf16*)(ws + 70 * MB);
    float*  Sb  = (float* )(ws + 102 * MB);
    __bf16* Pb  = (__bf16*)(ws + 166 * MB);

    // K1: weight convert+transpose (coalesced via LDS tile), all 3 in one grid
    wcvt3_kernel<<<dim3(32, 32, 3), 256, 0, stream>>>(Wq, Wk, Wv, WqT, WkT, WvT);

    // K2: projections over all 16384 rows (wave = 32 rows x 64 cols)
    dim3 gp(64, 16);
    proj_kernel<<<gp, 256, 0, stream>>>(x, WqT, Qb, nullptr);
    proj_kernel<<<gp, 256, 0, stream>>>(x, WkT, Kb, nullptr);
    proj_kernel<<<gp, 256, 0, stream>>>(x, WvT, nullptr, Vt);

    // Per batch: scores -> softmax -> weighted sum + GELU + residual
    for (int b = 0; b < NBATCH; ++b) {
        const __bf16* qb  = Qb + (size_t)b * SEQ * DIM;
        const __bf16* kb  = Kb + (size_t)b * SEQ * DIM;
        const __bf16* vtb = Vt + (size_t)b * DIM * SEQ;
        const float*  xb  = x   + (size_t)b * SEQ * DIM;
        float*        ob  = out + (size_t)b * SEQ * DIM;

        qk_kernel<<<dim3(16, 32), 256, 0, stream>>>(qb, kb, Sb);
        softmax_kernel<<<SEQ, 256, 0, stream>>>(Sb, Pb);
        pv_kernel<<<dim3(16, 16), 256, 0, stream>>>(Pb, vtb, xb, ob);
    }
}
